// GraphEmbedderMLP_13786845020220
// MI455X (gfx1250) — matrix operand order repaired and checked by reading, not compiled, not run on hardware
//
#include <hip/hip_runtime.h>
#include <hip/hip_bf16.h>

// ---------------- problem constants (match reference) ----------------
#define N_ENT   500000
#define N_REL   64
#define HID     256
#define EMB     128
#define E_EDGES 400000

typedef __attribute__((ext_vector_type(16))) __bf16 v16bf;
typedef __attribute__((ext_vector_type(8)))  __bf16 v8bf;
typedef __attribute__((ext_vector_type(8)))  float  v8f;
typedef __attribute__((ext_vector_type(4)))  float  v4f;

// LDS layout (bytes): W1relT 32K | W2T 128K | W3T 64K | staging 64K  = 288K
#define LDS_W1R_ELEMS (HID * N_REL)     // stored as [n][k], stride N_REL
#define LDS_W2T_ELEMS (HID * HID)       // stored as [n][k], stride HID
#define LDS_W3T_ELEMS (EMB * HID)       // stored as [n][k], stride HID
#define LDS_STG_ELEMS (8 * 16 * HID)    // 8 waves x 16 rows x HID, bf16
#define LDS_TOTAL_BYTES ((LDS_W1R_ELEMS + LDS_W2T_ELEMS + LDS_W3T_ELEMS + LDS_STG_ELEMS) * 2)

// ---- A-operand fragment (16x32 bf16), ISA "16-bit A-Matrix 16x32" layout:
//   lane < 16  : row = lane,    K = {k0+0..7,  k0+16..23}
//   lane >= 16 : row = lane-16, K = {k0+8..15, k0+24..31}
// `base` points at (row 0, k0) of a row-major matrix; `stride` in elements.
__device__ __forceinline__ v16bf ld_fragA_lds(const __bf16* base, int stride, int lane) {
    const int half = lane >> 4;
    const int r    = lane & 15;
    const __bf16* q = base + r * stride + half * 8;
    v8bf lo = *(const v8bf*)(q);        // K = k0 + half*8 + 0..7
    v8bf hi = *(const v8bf*)(q + 16);   // K = k0 + 16 + half*8 + 0..7
    v16bf out;
#pragma unroll
    for (int i = 0; i < 8; ++i) { out[i] = lo[i]; out[i + 8] = hi[i]; }
    return out;
}

// ---- B-operand fragment (32x16 bf16), same K order as the A fragment
// (lanes 0-15 hold K={0..7,16..23}, lanes 16-31 hold K={8..15,24..31},
//  N = lane&15).  Weights are stored
// transposed in LDS as [n][k], so each lane reads two 16-byte chunks.
__device__ __forceinline__ v16bf ld_fragB_lds(const __bf16* base, int stride, int lane) {
    const int half = lane >> 4;
    const int n    = lane & 15;
    const __bf16* q = base + n * stride + half * 8;
    v8bf lo = *(const v8bf*)(q);        // K = k0 + half*8 + 0..7
    v8bf hi = *(const v8bf*)(q + 16);   // K = k0 + 16 + half*8 + 0..7
    v16bf out;
#pragma unroll
    for (int i = 0; i < 8; ++i) { out[i] = lo[i]; out[i + 8] = hi[i]; }
    return out;
}

// A-fragment sourced from global fp32 (edge_attr), converted to bf16.
__device__ __forceinline__ v16bf ld_fragA_gf32(const float* base, int stride, int lane) {
    const int half = lane >> 4;
    const int r    = lane & 15;
    const float* q = base + r * stride + half * 8;
    v4f a0 = *(const v4f*)(q);
    v4f a1 = *(const v4f*)(q + 4);
    v4f b0 = *(const v4f*)(q + 16);
    v4f b1 = *(const v4f*)(q + 20);
    v16bf out;
#pragma unroll
    for (int i = 0; i < 4; ++i) {
        out[i]      = (__bf16)a0[i];
        out[i + 4]  = (__bf16)a1[i];
        out[i + 8]  = (__bf16)b0[i];
        out[i + 12] = (__bf16)b1[i];
    }
    return out;
}

__device__ __forceinline__ v8f wmma_bf16(v16bf a, v16bf b, v8f c) {
    // D = A(16x32 bf16) x B(32x16 bf16) + C(16x16 f32)
    return __builtin_amdgcn_wmma_f32_16x16x32_bf16(
        /*neg_a=*/false, a, /*neg_b=*/false, b,
        /*c_mod=*/(short)0, c, /*reuse_a=*/false, /*reuse_b=*/false);
}

__global__ void edge_mlp_kernel(const int*  __restrict__ edge_index,   // (2, E)
                                const float* __restrict__ edge_attr,   // (E, 64)
                                const float* __restrict__ W1,          // (N_ENT+64, 256)
                                const float* __restrict__ b1,          // (256,)
                                const float* __restrict__ W2,          // (256, 256)
                                const float* __restrict__ b2,          // (256,)
                                const float* __restrict__ W3,          // (256, 128)
                                const float* __restrict__ b3,          // (128,)
                                float* __restrict__ out,               // (2, E, 128)
                                int nTiles)                            // (2E)/128
{
    extern __shared__ __align__(16) char smem_raw[];
    __bf16* w1rT = (__bf16*)smem_raw;                 // [HID][N_REL]
    __bf16* w2T  = w1rT + LDS_W1R_ELEMS;              // [HID][HID]
    __bf16* w3T  = w2T  + LDS_W2T_ELEMS;              // [EMB][HID]
    __bf16* stg  = w3T  + LDS_W3T_ELEMS;              // [8][16][HID]

    const int tid = threadIdx.x;

    // ---- Stage weights into LDS once, transposed to [n][k] in bf16 ----
    for (int i = tid; i < N_REL * HID; i += blockDim.x) {
        int k = i / HID, n = i % HID;
        w1rT[n * N_REL + k] = (__bf16)W1[(size_t)(N_ENT + k) * HID + n];
    }
    for (int i = tid; i < HID * HID; i += blockDim.x) {
        int k = i / HID, n = i % HID;
        w2T[n * HID + k] = (__bf16)W2[i];
    }
    for (int i = tid; i < HID * EMB; i += blockDim.x) {
        int k = i / EMB, n = i % EMB;
        w3T[n * HID + k] = (__bf16)W3[i];
    }
    __syncthreads();

    const int wave = tid >> 5;
    const int lane = tid & 31;
    const int half = lane >> 4;
    const int ln   = lane & 15;
    __bf16* myStg = stg + wave * (16 * HID);   // wave-private 16-row staging

    // Persistent grid-stride over 128-row tiles (8 waves x 16 rows).
    for (int t = blockIdx.x; t < nTiles; t += gridDim.x) {
        const int rowBase = t * 128 + wave * 16;          // first of 16 rows
        const int s       = rowBase >= E_EDGES;           // side (0/1)
        const int eBase   = rowBase - s * E_EDGES;        // edge id of row 0

        // speculative prefetch of the next tile's edge_attr rows
        __builtin_prefetch(edge_attr + (size_t)((eBase + 128) % E_EDGES) * N_REL, 0, 1);

        // -------- Layer 1: h0 = relu(W1_ent[idx] + edge_attr @ W1_rel + b1)
        v16bf a0 = ld_fragA_gf32(edge_attr + (size_t)eBase * N_REL + 0,  N_REL, lane);
        v16bf a1 = ld_fragA_gf32(edge_attr + (size_t)eBase * N_REL + 32, N_REL, lane);

        int rowIdx[8];   // entity index for the 8 C-rows this lane owns
#pragma unroll
        for (int v = 0; v < 8; ++v)
            rowIdx[v] = edge_index[(size_t)s * E_EDGES + eBase + half * 8 + v];

        for (int nt = 0; nt < HID / 16; ++nt) {
            const int n0 = nt * 16;
            v8f acc = {};
            acc = wmma_bf16(a0, ld_fragB_lds(w1rT + n0 * N_REL + 0,  N_REL, lane), acc);
            acc = wmma_bf16(a1, ld_fragB_lds(w1rT + n0 * N_REL + 32, N_REL, lane), acc);
            const int n = n0 + ln;
            const float bias = b1[n];
#pragma unroll
            for (int v = 0; v < 8; ++v) {
                float x = acc[v] + W1[(size_t)rowIdx[v] * HID + n] + bias;
                x = x > 0.f ? x : 0.f;
                myStg[(half * 8 + v) * HID + n] = (__bf16)x;   // C-layout scatter
            }
        }

        // -------- Layer 2: h1 = relu(h0 @ W2 + b2) ----------------------
        v16bf h0f[8];
#pragma unroll
        for (int ks = 0; ks < 8; ++ks)
            h0f[ks] = ld_fragA_lds(myStg + ks * 32, HID, lane);  // full h0 in regs

        for (int nt = 0; nt < HID / 16; ++nt) {
            const int n0 = nt * 16;
            v8f acc = {};
#pragma unroll
            for (int ks = 0; ks < 8; ++ks)
                acc = wmma_bf16(h0f[ks], ld_fragB_lds(w2T + n0 * HID + ks * 32, HID, lane), acc);
            const int n = n0 + ln;
            const float bias = b2[n];
#pragma unroll
            for (int v = 0; v < 8; ++v) {
                float x = acc[v] + bias;
                x = x > 0.f ? x : 0.f;
                myStg[(half * 8 + v) * HID + n] = (__bf16)x;   // overwrite h0 (already in regs)
            }
        }

        // -------- Layer 3: out = h1 @ W3 + b3 ---------------------------
        v16bf h1f[8];
#pragma unroll
        for (int ks = 0; ks < 8; ++ks)
            h1f[ks] = ld_fragA_lds(myStg + ks * 32, HID, lane);

        float* outBase = out + (size_t)rowBase * EMB;
        for (int nt = 0; nt < EMB / 16; ++nt) {
            const int n0 = nt * 16;
            v8f acc = {};
#pragma unroll
            for (int ks = 0; ks < 8; ++ks)
                acc = wmma_bf16(h1f[ks], ld_fragB_lds(w3T + n0 * HID + ks * 32, HID, lane), acc);
            const int n = n0 + ln;
            const float bias = b3[n];
#pragma unroll
            for (int v = 0; v < 8; ++v)
                outBase[(half * 8 + v) * EMB + n] = acc[v] + bias;
        }
    }
}

extern "C" void kernel_launch(void* const* d_in, const int* in_sizes, int n_in,
                              void* d_out, int out_size, void* d_ws, size_t ws_size,
                              hipStream_t stream) {
    const int*   edge_index = (const int*)  d_in[0];
    const float* edge_attr  = (const float*)d_in[1];
    const float* W1         = (const float*)d_in[2];
    const float* b1         = (const float*)d_in[3];
    const float* W2         = (const float*)d_in[4];
    const float* b2         = (const float*)d_in[5];
    const float* W3         = (const float*)d_in[6];
    const float* b3         = (const float*)d_in[7];
    float* out = (float*)d_out;

    const int nTiles = (2 * E_EDGES) / 128;   // 6250, exact
    const size_t shmem = (size_t)LDS_TOTAL_BYTES;  // 288 KB < 320 KB/WGP

    hipFuncSetAttribute((const void*)edge_mlp_kernel,
                        hipFuncAttributeMaxDynamicSharedMemorySize, (int)shmem);

    int grid = nTiles < 1024 ? nTiles : 1024;  // persistent workgroups
    edge_mlp_kernel<<<grid, 256, shmem, stream>>>(
        edge_index, edge_attr, W1, b1, W2, b2, W3, b3, out, nTiles);
}
